// RRN_83958020702600
// MI455X (gfx1250) — matrix operand order repaired and checked by reading, not compiled, not run on hardware
//
#include <hip/hip_runtime.h>

#define HID    96
#define NNODE  1296    // 16 boards * 81 cells
#define NEDGE  25920   // 16 boards * 1620 edges
#define NSTEPS 32

typedef __attribute__((ext_vector_type(16))) _Float16 v16h;
typedef __attribute__((ext_vector_type(8)))  _Float16 v8h;
typedef __attribute__((ext_vector_type(8)))  float    v8f;

__device__ __forceinline__ float sigmoidf_(float x) { return 1.0f / (1.0f + __expf(-x)); }

// 20 fixed Sudoku neighbors of cell dst: 8 row peers, 8 col peers, 4 block-only peers.
__device__ __forceinline__ int nbr_cell(int dst, int j) {
  int r = dst / 9, c = dst - r * 9;
  if (j < 8)  { int cc = j + (j >= c); return r * 9 + cc; }
  if (j < 16) { int k = j - 8; int rr = k + (k >= r); return rr * 9 + c; }
  int k  = j - 16;
  int br = (r / 3) * 3, bc = (c / 3) * 3;
  int lr = r - br,      lc = c - bc;
  int sr = (k >> 1); sr += (sr >= lr);
  int sc = (k & 1);  sc += (sc >= lc);
  return (br + sr) * 9 + (bc + sc);
}

// A fragment (16x32 f16, M x K). lanes 0-15 = rows, hi bit selects K-halves:
// element e -> K = kbase + hi*16 + e (contiguous, matches packed B)
template<int KSTEPS>
__device__ __forceinline__ void load_a_frags(v16h* a, const _Float16* As, int lda,
                                             int mrow, int lane) {
  int hi = lane >> 4;
  #pragma unroll
  for (int ks = 0; ks < KSTEPS; ++ks) {
    const _Float16* p = As + mrow * lda + ks * 32 + hi * 16;
    v8h lo = *(const v8h*)p;
    v8h hv = *(const v8h*)(p + 8);
    v16h f;
    #pragma unroll
    for (int i = 0; i < 8; ++i) { f[i] = lo[i]; f[i + 8] = hv[i]; }
    a[ks] = f;
  }
}

__device__ __forceinline__ v16h load_a_one(const _Float16* As, int lda, int mrow,
                                           int kbase, int lane) {
  int hi = lane >> 4;
  const _Float16* p = As + mrow * lda + kbase + hi * 16;
  v8h lo = *(const v8h*)p;
  v8h hv = *(const v8h*)(p + 8);
  v16h f;
  #pragma unroll
  for (int i = 0; i < 8; ++i) { f[i] = lo[i]; f[i + 8] = hv[i]; }
  return f;
}

// Pre-packed B fragment: fragment `frag` = ks*NTILES + nt; each lane's 16 halves are
// 32 contiguous bytes at P[((frag*32 + lane) * 16)].  -> two global_load_b128.
__device__ __forceinline__ v16h load_b_packed(const _Float16* __restrict__ P, int frag,
                                              int lane) {
  const _Float16* p = P + (((frag << 5) + lane) << 4);
  v8h lo = *(const v8h*)p;
  v8h hv = *(const v8h*)(p + 8);
  v16h f;
  #pragma unroll
  for (int i = 0; i < 8; ++i) { f[i] = lo[i]; f[i + 8] = hv[i]; }
  return f;
}

// Register-blocked multi-M-tile GEMM layer.  Wave owns MT 16-row tiles starting at mbase.
// N dimension is processed in chunks of NTC tiles to bound live accumulators:
// live = MT*NTC acc (72 VGPRs at 3x3) + MT A-frags (24) + 1 B-frag (8)  -> no spills.
// Each B fragment still feeds MT WMMAs (3x less weight traffic at MT=3).
template<int MT, int KS, int NT, int NTC, bool RELU, bool TO_GLOBAL>
__device__ __forceinline__ void gemm_multi(const _Float16* As, int lda,
                                           const _Float16* __restrict__ Pw,
                                           const float* bias,
                                           _Float16* Os, int ldo,
                                           float* Og, int ldog, int growbase,
                                           int mbase, int lane) {
  int n = lane & 15, hi = lane >> 4;
  v8f zero = {};
  #pragma unroll
  for (int nt0 = 0; nt0 < NT; nt0 += NTC) {
    v8f acc[MT * NTC];
    #pragma unroll
    for (int i = 0; i < MT * NTC; ++i) acc[i] = zero;

    #pragma unroll
    for (int ks = 0; ks < KS; ++ks) {
      v16h a[MT];
      #pragma unroll
      for (int mt = 0; mt < MT; ++mt)
        a[mt] = load_a_one(As, lda, mbase + mt * 16 + n, ks * 32, lane);
      #pragma unroll
      for (int c = 0; c < NTC; ++c) {
        v16h bf = load_b_packed(Pw, ks * NT + nt0 + c, lane);
        #pragma unroll
        for (int mt = 0; mt < MT; ++mt)
          acc[c * MT + mt] = __builtin_amdgcn_wmma_f32_16x16x32_f16(
              false, a[mt], false, bf, (short)0, acc[c * MT + mt], false, false);
      }
    }

    #pragma unroll
    for (int c = 0; c < NTC; ++c) {
      float bv = bias[(nt0 + c) * 16 + n];
      #pragma unroll
      for (int mt = 0; mt < MT; ++mt) {
        #pragma unroll
        for (int r = 0; r < 8; ++r) {
          float v = acc[c * MT + mt][r] + bv;
          if (RELU) v = fmaxf(v, 0.0f);
          int row = mbase + mt * 16 + hi * 8 + r;
          if (TO_GLOBAL) Og[(size_t)(growbase + row) * ldog + (nt0 + c) * 16 + n] = v;
          else           Os[row * ldo + (nt0 + c) * 16 + n] = (_Float16)v;
        }
      }
    }
  }
}

// Single-tile variant (used by the node kernel).
template<int KSTEPS, int NTILES, bool RELU>
__device__ __forceinline__ void gemm_to_smem(const v16h* a, const _Float16* __restrict__ Pw,
                                             const float* bias, _Float16* Os, int ldo,
                                             int mbase, int lane) {
  int n = lane & 15, hi = lane >> 4;
  #pragma unroll
  for (int nt = 0; nt < NTILES; ++nt) {
    v8f acc = {};
    #pragma unroll
    for (int ks = 0; ks < KSTEPS; ++ks) {
      v16h bf = load_b_packed(Pw, ks * NTILES + nt, lane);
      acc = __builtin_amdgcn_wmma_f32_16x16x32_f16(false, a[ks], false, bf,
                                                   (short)0, acc, false, false);
    }
    float bv = bias[nt * 16 + n];
    #pragma unroll
    for (int r = 0; r < 8; ++r) {
      float v = acc[r] + bv;
      if (RELU) v = fmaxf(v, 0.0f);
      Os[(mbase + hi * 8 + r) * ldo + nt * 16 + n] = (_Float16)v;
    }
  }
}

// ---------------- one-time kernels ----------------

// Pack a row-major KxN f32 weight into WMMA-B fragment layout (f16).
// i = {ks, nt, lane, e}: dst[i] = W[(ks*32 + (lane>>4)*16 + e) * N + nt*16 + (lane&15)]
__device__ __forceinline__ void pack_frag(_Float16* dst, const float* src, int N,
                                          int ntiles, int i) {
  int e    = i & 15;
  int lane = (i >> 4) & 31;
  int rest = i >> 9;
  int nt   = rest % ntiles;
  int ks   = rest / ntiles;
  int k = ks * 32 + ((lane >> 4) << 4) + e;
  int n = nt * 16 + (lane & 15);
  dst[i] = (_Float16)src[k * N + n];
}

__global__ void rrn_weights(const float* mw1, const float* mw2, const float* mw3,
                            const float* lw1, const float* lw2, const float* lw3,
                            const float* Wih, const float* Whh, const float* Wout,
                            _Float16* pmW1, _Float16* pmW2, _Float16* pmW3,
                            _Float16* plW1, _Float16* plW2, _Float16* plW3,
                            _Float16* pGate, _Float16* pWout) {
  int i = blockIdx.x * 256 + threadIdx.x;   // grid covers 192*384 = 73728
  if (i < 192 * 96) { pack_frag(pmW1, mw1, 96, 6, i); pack_frag(plW1, lw1, 96, 6, i); }
  if (i < 96 * 96)  { pack_frag(pmW2, mw2, 96, 6, i); pack_frag(pmW3, mw3, 96, 6, i);
                      pack_frag(plW2, lw2, 96, 6, i); pack_frag(plW3, lw3, 96, 6, i); }
  if (i < 96 * 384) {
    // combined gate weight: rows 0:96 = W_ih^T, rows 96:192 = W_hh^T  (K=192, N=384)
    int e = i & 15, lane = (i >> 4) & 31, rest = i >> 9;
    int nt = rest % 24, ks = rest / 24;
    int k = ks * 32 + ((lane >> 4) << 4) + e;
    int g = nt * 16 + (lane & 15);
    float v = (k < 96) ? Wih[g * 96 + k] : Whh[g * 96 + (k - 96)];
    pGate[i] = (_Float16)v;
  }
  if (i < 96 * 16) {
    int e = i & 15, lane = (i >> 4) & 31, ks = i >> 9;   // single ntile
    int k = ks * 32 + ((lane >> 4) << 4) + e;
    int n = lane & 15;
    pWout[i] = (n < 10) ? (_Float16)Wout[k * 10 + n] : (_Float16)0.0f;
  }
}

// e2x MLP (one-hot input => row gather of W1) + state init. 1 block per node, 96 threads.
__global__ void rrn_init(const int* __restrict__ board,
                         const float* W1, const float* b1,
                         const float* W2, const float* b2,
                         const float* W3, const float* b3,
                         _Float16* xf, _Float16* hmsg, _Float16* hbuf, float* cbuf) {
  __shared__ float s0[96], s1[96];
  int n = blockIdx.x;            // global node 0..1295
  int t = threadIdx.x;           // 0..95
  int cell = n % 81;
  int r = cell / 9, cc = cell - r * 9;
  int d = board[n];
  float h1 = W1[d * 96 + t] + W1[(16 + r) * 96 + t] + W1[(32 + cc) * 96 + t] + b1[t];
  s0[t] = fmaxf(h1, 0.0f);
  __syncthreads();
  float h2 = b2[t];
  for (int k = 0; k < 96; ++k) h2 += s0[k] * W2[k * 96 + t];
  s1[t] = fmaxf(h2, 0.0f);
  __syncthreads();
  float x = b3[t];
  for (int k = 0; k < 96; ++k) x += s1[k] * W3[k * 96 + t];
  _Float16 xh = (_Float16)x;
  xf[n * 96 + t]   = xh;
  hmsg[n * 96 + t] = xh;          // h_msg init = x
  hbuf[n * 96 + t] = (_Float16)0.0f;
  cbuf[n * 96 + t] = 0.0f;
}

// ---------------- per-step kernels ----------------

// Message MLP: block = 128 threads = 4 waves, 192 edges; each wave owns 48 rows (3 M tiles).
__global__ void __launch_bounds__(128, 1)
rrn_msg(const _Float16* __restrict__ hmsg,
        const _Float16* __restrict__ pW1, const _Float16* __restrict__ pW2,
        const _Float16* __restrict__ pW3,
        const float* b1, const float* b2, const float* b3,
        float* __restrict__ msg) {
  extern __shared__ char smem[];
  _Float16* sA = (_Float16*)smem;    // 192 x 192
  _Float16* sB = sA + 192 * 192;     // 192 x 96
  int tid  = threadIdx.x;            // 128
  int lane = tid & 31, wave = tid >> 5;
  int mbase = wave * 48;

  // stage pair = [h[src] | h[dst]]; edge id = b*1620 + dst*20 + j
  for (int idx = tid; idx < 192 * 2; idx += 128) {
    int le = idx >> 1, half = idx & 1;
    int ge = blockIdx.x * 192 + le;
    int b  = ge / 1620; int r0 = ge - b * 1620;
    int dc = r0 / 20;   int j  = r0 - dc * 20;
    int cell = half ? dc : nbr_cell(dc, j);
    int gn = b * 81 + cell;
    const uint4* src = (const uint4*)(hmsg + gn * 96);
    uint4* dst = (uint4*)(sA + le * 192 + half * 96);
    #pragma unroll
    for (int i = 0; i < 12; ++i) dst[i] = src[i];
  }
  __syncthreads();
  // All subsequent LDS traffic stays within this wave's 48-row slab -> no more barriers.
  gemm_multi<3, 6, 6, 3, true,  false>(sA, 192, pW1, b1, sB, 96, nullptr, 0, 0, mbase, lane);
  gemm_multi<3, 3, 6, 3, true,  false>(sB,  96, pW2, b2, sA, 192, nullptr, 0, 0, mbase, lane);
  gemm_multi<3, 3, 6, 3, false, true >(sA, 192, pW3, b3, nullptr, 0, msg, 96,
                                       blockIdx.x * 192, mbase, lane);
}

// Node update over 48 nodes per block (3 waves): agg -> lin MLP -> gates -> LSTM -> out head.
__global__ void __launch_bounds__(96, 1)
rrn_node(const float* __restrict__ msg, const _Float16* __restrict__ xf,
         const _Float16* __restrict__ pW1, const _Float16* __restrict__ pW2,
         const _Float16* __restrict__ pW3,
         const float* lb1, const float* lb2, const float* lb3,
         const _Float16* __restrict__ pGate,
         const float* bih, const float* bhh,
         const _Float16* __restrict__ pWout, const float* bout,
         _Float16* hmsg, _Float16* hbuf, float* cbuf,
         float* __restrict__ out, int step) {
  extern __shared__ char smem[];
  _Float16* sA = (_Float16*)smem;        // 48 x 192
  _Float16* sB = sA + 48 * 192;          // 48 x 96
  float*    sG = (float*)(sB + 48 * 96); // 48 x 384 gates (f32)

  int tid  = threadIdx.x;                // 96
  int lane = tid & 31, wave = tid >> 5;
  int mbase = wave * 16;
  int base  = blockIdx.x * 48;

  // A = [agg | x]; agg = sum of the node's 20 contiguous message rows
  for (int idx = tid; idx < 48 * 192; idx += 96) {
    int m = idx / 192, col = idx - m * 192;
    int gm = base + m;
    if (col < 96) {
      const float* mp = msg + (size_t)(20 * gm) * 96 + col;
      float s = 0.0f;
      #pragma unroll
      for (int u = 0; u < 20; ++u) s += mp[u * 96];
      sA[idx] = (_Float16)s;
    } else {
      sA[idx] = xf[gm * 96 + (col - 96)];
    }
  }
  __syncthreads();
  { v16h a[6]; load_a_frags<6>(a, sA, 192, mbase + (lane & 15), lane);
    gemm_to_smem<6, 6, true>(a, pW1, lb1, sB, 96, mbase, lane); }
  __syncthreads();
  { v16h a[3]; load_a_frags<3>(a, sB, 96, mbase + (lane & 15), lane);
    gemm_to_smem<3, 6, true>(a, pW2, lb2, sA, 192, mbase, lane); }
  __syncthreads();
  { v16h a[3]; load_a_frags<3>(a, sA, 192, mbase + (lane & 15), lane);
    gemm_to_smem<3, 6, false>(a, pW3, lb3, sB, 96, mbase, lane); }  // lstm_in -> sB
  __syncthreads();

  // gate input = [lstm_in | h_old]
  for (int idx = tid; idx < 48 * 192; idx += 96) {
    int m = idx / 192, col = idx - m * 192;
    sA[idx] = (col < 96) ? sB[m * 96 + col] : hbuf[(base + m) * 96 + (col - 96)];
  }
  __syncthreads();

  // gates = [lstm_in | h_old] @ pGate  (K=192, N=384, 24 ntiles), all B frags pre-packed
  {
    v16h ag[6];
    load_a_frags<6>(ag, sA, 192, mbase + (lane & 15), lane);
    int ncl = lane & 15, hib = lane >> 4;
    #pragma unroll 4
    for (int nt = 0; nt < 24; ++nt) {
      v8f acc = {};
      #pragma unroll
      for (int ks = 0; ks < 6; ++ks) {
        v16h bf = load_b_packed(pGate, ks * 24 + nt, lane);
        acc = __builtin_amdgcn_wmma_f32_16x16x32_f16(false, ag[ks], false, bf,
                                                     (short)0, acc, false, false);
      }
      int ncol = nt * 16 + ncl;
      float bv = bih[ncol] + bhh[ncol];
      #pragma unroll
      for (int r = 0; r < 8; ++r)
        sG[(mbase + hib * 8 + r) * 384 + ncol] = acc[r] + bv;
    }
  }
  __syncthreads();

  // LSTM elementwise: gates order [i | f | g | o]
  for (int idx = tid; idx < 48 * 96; idx += 96) {
    int m = idx / 96, col = idx - m * 96;
    int gm = base + m;
    float ig = sG[m * 384 + col];
    float fg = sG[m * 384 + 96 + col];
    float gg = sG[m * 384 + 192 + col];
    float og = sG[m * 384 + 288 + col];
    float cold = cbuf[gm * 96 + col];
    float cn = sigmoidf_(fg) * cold + sigmoidf_(ig) * tanhf(gg);
    float hn = sigmoidf_(og) * tanhf(cn);
    cbuf[gm * 96 + col] = cn;
    _Float16 hh = (_Float16)hn;
    hbuf[gm * 96 + col] = hh;
    hmsg[gm * 96 + col] = hh;
    sB[m * 96 + col] = hh;
  }
  __syncthreads();

  // output head: h_new @ pWout (96x16, cols 10..15 zero)
  {
    int ncl = lane & 15, hib = lane >> 4;
    v16h a[3];
    load_a_frags<3>(a, sB, 96, mbase + (lane & 15), lane);
    v8f acc = {};
    #pragma unroll
    for (int ks = 0; ks < 3; ++ks) {
      v16h bf = load_b_packed(pWout, ks, lane);
      acc = __builtin_amdgcn_wmma_f32_16x16x32_f16(false, a[ks], false, bf,
                                                   (short)0, acc, false, false);
    }
    if (ncl < 10) {
      float bv = bout[ncl];
      #pragma unroll
      for (int r = 0; r < 8; ++r) {
        int gm = base + mbase + hib * 8 + r;
        out[(size_t)step * (NNODE * 10) + gm * 10 + ncl] = acc[r] + bv;
      }
    }
  }
}

// ---------------- host launcher ----------------

extern "C" void kernel_launch(void* const* d_in, const int* in_sizes, int n_in,
                              void* d_out, int out_size, void* d_ws, size_t ws_size,
                              hipStream_t stream) {
  (void)in_sizes; (void)n_in; (void)out_size; (void)ws_size;

  const int*   board = (const int*)d_in[0];
  const float* eW1 = (const float*)d_in[1];  const float* eb1 = (const float*)d_in[2];
  const float* eW2 = (const float*)d_in[3];  const float* eb2 = (const float*)d_in[4];
  const float* eW3 = (const float*)d_in[5];  const float* eb3 = (const float*)d_in[6];
  const float* mw1 = (const float*)d_in[7];  const float* mb1 = (const float*)d_in[8];
  const float* mw2 = (const float*)d_in[9];  const float* mb2 = (const float*)d_in[10];
  const float* mw3 = (const float*)d_in[11]; const float* mb3 = (const float*)d_in[12];
  const float* lw1 = (const float*)d_in[13]; const float* lb1 = (const float*)d_in[14];
  const float* lw2 = (const float*)d_in[15]; const float* lb2 = (const float*)d_in[16];
  const float* lw3 = (const float*)d_in[17]; const float* lb3 = (const float*)d_in[18];
  const float* Wih = (const float*)d_in[19]; const float* Whh = (const float*)d_in[20];
  const float* bih = (const float*)d_in[21]; const float* bhh = (const float*)d_in[22];
  const float* Wou = (const float*)d_in[23]; const float* bou = (const float*)d_in[24];

  char* w = (char*)d_ws;
  auto take = [&](size_t bytes) { char* p = w; w += bytes; return p; };
  float*    msgbuf = (float*)take((size_t)NEDGE * 96 * 4);   // 9.95 MB
  _Float16* xf     = (_Float16*)take((size_t)NNODE * 96 * 2);
  _Float16* hmsg   = (_Float16*)take((size_t)NNODE * 96 * 2);
  _Float16* hbuf   = (_Float16*)take((size_t)NNODE * 96 * 2);
  float*    cbuf   = (float*)take((size_t)NNODE * 96 * 4);
  _Float16* pmW1   = (_Float16*)take(192 * 96 * 2);
  _Float16* pmW2   = (_Float16*)take(96 * 96 * 2);
  _Float16* pmW3   = (_Float16*)take(96 * 96 * 2);
  _Float16* plW1   = (_Float16*)take(192 * 96 * 2);
  _Float16* plW2   = (_Float16*)take(96 * 96 * 2);
  _Float16* plW3   = (_Float16*)take(96 * 96 * 2);
  _Float16* pGate  = (_Float16*)take(192 * 384 * 2);
  _Float16* pWout  = (_Float16*)take(96 * 16 * 2);

  rrn_weights<<<dim3(288), dim3(256), 0, stream>>>(
      mw1, mw2, mw3, lw1, lw2, lw3, Wih, Whh, Wou,
      pmW1, pmW2, pmW3, plW1, plW2, plW3, pGate, pWout);

  rrn_init<<<dim3(NNODE), dim3(96), 0, stream>>>(
      board, eW1, eb1, eW2, eb2, eW3, eb3, xf, hmsg, hbuf, cbuf);

  const size_t smem_msg  = (192 * 192 + 192 * 96) * sizeof(_Float16);   // 110592
  const size_t smem_node = (48 * 192 + 48 * 96) * sizeof(_Float16)
                         + 48 * 384 * sizeof(float);                    // 101376

  for (int t = 0; t < NSTEPS; ++t) {
    rrn_msg<<<dim3(NEDGE / 192), dim3(128), smem_msg, stream>>>(
        hmsg, pmW1, pmW2, pmW3, mb1, mb2, mb3, msgbuf);
    rrn_node<<<dim3(NNODE / 48), dim3(96), smem_node, stream>>>(
        msgbuf, xf, plW1, plW2, plW3, lb1, lb2, lb3,
        pGate, bih, bhh, pWout, bou,
        hmsg, hbuf, cbuf, (float*)d_out, t);
  }
}
